// ODENetManual_41085657153954
// MI455X (gfx1250) — matrix operand order repaired and checked by reading, not compiled, not run on hardware
//
#include <hip/hip_runtime.h>

// ---------------------------------------------------------------------------
// Types for CDNA5 WMMA (gfx1250, wave32)
// ---------------------------------------------------------------------------
typedef __attribute__((ext_vector_type(16))) __bf16          v16bf;
typedef __attribute__((ext_vector_type(8)))  float           v8f;
typedef __attribute__((ext_vector_type(8)))  unsigned short  u16x8;

union BfFrag { v16bf v; u16x8 h[2]; };

__device__ __forceinline__ unsigned short f2bf(float f) {
  unsigned int u = __builtin_bit_cast(unsigned int, f);
  unsigned int rnd = 0x7FFFu + ((u >> 16) & 1u);
  u += rnd;
  return (unsigned short)(u >> 16);
}

// ---------------------------------------------------------------------------
// Stem conv: x [64,1,64,64] fp32 -> out NHWC fp32 [64,62,62,64], pad 0, +bias
// ---------------------------------------------------------------------------
__global__ void stem_kernel(const float* __restrict__ x,
                            const float* __restrict__ w,
                            const float* __restrict__ bias,
                            float* __restrict__ out) {
  int idx = blockIdx.x * blockDim.x + threadIdx.x;
  const int total = 64 * 62 * 62 * 64;
  if (idx >= total) return;
  int co = idx & 63;
  int t  = idx >> 6;
  int xo = t % 62; t /= 62;
  int yo = t % 62;
  int b  = t / 62;
  const float* xb = x + (size_t)b * 64 * 64;
  float acc = bias[co];
  #pragma unroll
  for (int dy = 0; dy < 3; ++dy)
    #pragma unroll
    for (int dx = 0; dx < 3; ++dx)
      acc += xb[(yo + dy) * 64 + (xo + dx)] * w[co * 9 + dy * 3 + dx];
  out[idx] = acc;
}

// ---------------------------------------------------------------------------
// GroupNorm (32 groups of 2 channels) + optional ReLU; writes fp32 and/or
// bf16(NHWC) outputs. One block per (batch, group). Input NHWC fp32, C=64.
// ---------------------------------------------------------------------------
__global__ void __launch_bounds__(128)
gn_kernel(const float* __restrict__ in,
          const float* __restrict__ gamma,
          const float* __restrict__ beta,
          float* __restrict__ outF,
          unsigned short* __restrict__ outB,
          int HW, int doRelu) {
  const int b  = blockIdx.x >> 5;
  const int g  = blockIdx.x & 31;
  const int c0 = g * 2;
  const int tid = threadIdx.x;
  const float* base = in + (size_t)b * HW * 64;

  float s = 0.f, ss = 0.f;
  for (int i = tid; i < HW; i += 128) {
    float x0 = base[(size_t)i * 64 + c0];
    float x1 = base[(size_t)i * 64 + c0 + 1];
    s += x0 + x1; ss += x0 * x0 + x1 * x1;
  }
  __shared__ float sh0[128];
  __shared__ float sh1[128];
  sh0[tid] = s; sh1[tid] = ss;
  __syncthreads();
  for (int o = 64; o > 0; o >>= 1) {
    if (tid < o) { sh0[tid] += sh0[tid + o]; sh1[tid] += sh1[tid + o]; }
    __syncthreads();
  }
  const float n    = (float)HW * 2.0f;
  const float mean = sh0[0] / n;
  const float var  = sh1[0] / n - mean * mean;
  const float rstd = rsqrtf(var + 1e-5f);
  const float g0 = gamma[c0] * rstd,     g1 = gamma[c0 + 1] * rstd;
  const float b0 = beta[c0] - mean * g0, b1 = beta[c0 + 1] - mean * g1;

  for (int i = tid; i < HW; i += 128) {
    float x0 = base[(size_t)i * 64 + c0]     * g0 + b0;
    float x1 = base[(size_t)i * 64 + c0 + 1] * g1 + b1;
    if (doRelu) { x0 = fmaxf(x0, 0.f); x1 = fmaxf(x1, 0.f); }
    size_t o = (size_t)b * HW * 64 + (size_t)i * 64 + c0;
    if (outF) { outF[o] = x0; outF[o + 1] = x1; }
    if (outB) { outB[o] = f2bf(x0); outB[o + 1] = f2bf(x1); }
  }
}

// ---------------------------------------------------------------------------
// Weight pack: fp32 [co][cinTotal][taps] (OIHW flat) -> bf16 [tap][co][ci<64];
// optional t-channel (ci==64) weights to fp32 twOut[tap][co].
// ---------------------------------------------------------------------------
__global__ void pack_w_kernel(const float* __restrict__ w,
                              unsigned short* __restrict__ wpk,
                              float* __restrict__ twOut,
                              int cinTotal, int taps) {
  int idx = blockIdx.x * blockDim.x + threadIdx.x;
  int total = taps * 64 * 64;
  if (idx < total) {
    int ci  = idx & 63;
    int co  = (idx >> 6) & 63;
    int tap = idx >> 12;
    wpk[idx] = f2bf(w[((size_t)co * cinTotal + ci) * taps + tap]);
  }
  if (twOut && idx < taps * 64) {
    int co  = idx & 63;
    int tap = idx >> 6;
    twOut[tap * 64 + co] = w[((size_t)co * cinTotal + 64) * taps + tap];
  }
}

// ---------------------------------------------------------------------------
// Implicit-GEMM conv via V_WMMA_F32_16X16X32_BF16 — all geometry compile-time.
//   inp: bf16 NHWC [64,Hi,Wi,64];  wpk: bf16 [TAPS][co(64)][ci(64)]
//   out: fp32 NHWC [64,Ho,Wo,64];  addres: optional residual (same shape)
//   tw/tval: optional concat-t channel contribution (t * sum of valid taps)
// Block = 128 threads (4 waves); wave w owns output channels [16w,16w+16).
// Each wave computes MT=2 adjacent 16-row M tiles sharing B fragments.
// Grid = ceil(64*Ho*Wo / (16*MT)).
// ---------------------------------------------------------------------------
template <int Hi, int Wi, int Ho, int Wo, int STRIDE, int PAD, int TAPS>
__global__ void __launch_bounds__(128)
conv_wmma_kernel(const unsigned short* __restrict__ inp,
                 const unsigned short* __restrict__ wpk,
                 float* __restrict__ out,
                 const float* __restrict__ addres,
                 const float* __restrict__ tw, float tval) {
  constexpr int MT = 2;
  constexpr int M  = 64 * Ho * Wo;
  const int tb   = blockIdx.x * MT;            // first 16-row tile index
  const int wave = threadIdx.x >> 5;
  const int lane = threadIdx.x & 31;
  const int n0   = wave * 16;
  const bool laneHi = lane >= 16;
  const int hi8c = laneHi ? 1 : 0;             // A channel-chunk offset (hi8/8)
  const int kloC = laneHi ? 1 : 0;             // B channel-chunk offset (klo/8)
  const int co   = n0 + (lane & 15);           // C/D column for this lane

  // Per-tile A-fragment row coordinates (lanes l and l+16 share M=l%16)
  int  ax[MT], ay[MT], ab[MT];
  bool mv[MT];
  #pragma unroll
  for (int j = 0; j < MT; ++j) {
    const int mA = (tb + j) * 16 + (lane & 15);
    mv[j] = mA < M;
    const int xx = mA % Wo;
    const int r  = mA / Wo;
    ax[j] = xx; ay[j] = r % Ho; ab[j] = r / Ho;
  }

  v8f acc[MT];
  #pragma unroll
  for (int j = 0; j < MT; ++j) acc[j] = (v8f){0.f,0.f,0.f,0.f,0.f,0.f,0.f,0.f};
  const u16x8 zer = {0, 0, 0, 0, 0, 0, 0, 0};

  #pragma unroll
  for (int tap = 0; tap < TAPS; ++tap) {
    constexpr int three = 3;
    const int dy = (TAPS == 1) ? 0 : tap / three;
    const int dx = (TAPS == 1) ? 0 : tap % three;

    // B fragments for this tap (shared across the MT tiles)
    const u16x8* wq = (const u16x8*)(wpk + ((size_t)(tap * 64 + co)) * 64);
    BfFrag bLo, bHi;
    bLo.h[0] = wq[kloC];         // ci = klo   .. klo+7
    bLo.h[1] = wq[kloC + 2];     // ci = 16+klo.. +7
    bHi.h[0] = wq[4 + kloC];     // ci = 32+klo.. +7
    bHi.h[1] = wq[6 + kloC];     // ci = 48+klo.. +7

    #pragma unroll
    for (int j = 0; j < MT; ++j) {
      const int yi = ay[j] * STRIDE + dy - PAD;
      const int xi = ax[j] * STRIDE + dx - PAD;
      const bool valid = mv[j] && yi >= 0 && yi < Hi && xi >= 0 && xi < Wi;

      BfFrag aLo, aHi;
      aLo.h[0] = zer; aLo.h[1] = zer; aHi.h[0] = zer; aHi.h[1] = zer;
      if (valid) {
        const u16x8* p = (const u16x8*)(inp + (((size_t)ab[j] * Hi + yi) * Wi + xi) * 64);
        aLo.h[0] = p[hi8c];        // ci = hi8   .. hi8+7
        aLo.h[1] = p[2 + hi8c];    // ci = 16+hi8.. +7
        aHi.h[0] = p[4 + hi8c];    // ci = 32+hi8.. +7
        aHi.h[1] = p[6 + hi8c];    // ci = 48+hi8.. +7
      }
      acc[j] = __builtin_amdgcn_wmma_f32_16x16x32_bf16(false, aLo.v, false, bLo.v,
                                                       (short)0, acc[j], false, false);
      acc[j] = __builtin_amdgcn_wmma_f32_16x16x32_bf16(false, aHi.v, false, bHi.v,
                                                       (short)0, acc[j], false, false);
    }
  }

  // Epilogue. D layout: lane col = co; VGPR v -> row tile*16 + v + (laneHi?8:0)
  #pragma unroll
  for (int j = 0; j < MT; ++j) {
    const int mBase = (tb + j) * 16 + (laneHi ? 8 : 0);
    #pragma unroll
    for (int v = 0; v < 8; ++v) {
      const int m = mBase + v;
      if (m < M) {
        float val = acc[j][v];
        const size_t oi = (size_t)m * 64 + co;
        if (addres) val += addres[oi];
        if (tw) {
          const int x = m % Wo;
          const int r = m / Wo;
          const int y = r % Ho;
          float s = 0.f;
          #pragma unroll
          for (int tap = 0; tap < 9; ++tap) {
            const int dy = tap / 3, dx = tap % 3;
            const int yi = y * STRIDE + dy - PAD;
            const int xi = x * STRIDE + dx - PAD;
            if (yi >= 0 && yi < Hi && xi >= 0 && xi < Wi) s += tw[tap * 64 + co];
          }
          val += tval * s;
        }
        out[oi] = val;
      }
    }
  }
}

// ---------------------------------------------------------------------------
// out = c0*a0 + c1*a1 [+ c2*a2 + c3*a3 + c4*a4]   (elementwise, alias-safe)
// ---------------------------------------------------------------------------
__global__ void lincomb_kernel(float* __restrict__ out,
                               const float* __restrict__ a0, float c0,
                               const float* __restrict__ a1, float c1,
                               const float* a2, float c2,
                               const float* a3, float c3,
                               const float* a4, float c4, int n) {
  int i = blockIdx.x * blockDim.x + threadIdx.x;
  if (i >= n) return;
  float v = c0 * a0[i] + c1 * a1[i];
  if (a2) v += c2 * a2[i];
  if (a3) v += c3 * a3[i];
  if (a4) v += c4 * a4[i];
  out[i] = v;
}

// ---------------------------------------------------------------------------
// Global average pool over 16x16 (NHWC fp32) -> pooled[b][c]
// ---------------------------------------------------------------------------
__global__ void pool_kernel(const float* __restrict__ in, float* __restrict__ pooled) {
  int b = blockIdx.x;
  int c = threadIdx.x;  // 64 threads
  float s = 0.f;
  for (int i = 0; i < 256; ++i) s += in[((size_t)b * 256 + i) * 64 + c];
  pooled[b * 64 + c] = s * (1.0f / 256.0f);
}

// ---------------------------------------------------------------------------
// FC: out[b][j] = pooled[b]·fcw[j] + fcb[j]  (64x10)
// ---------------------------------------------------------------------------
__global__ void fc_kernel(const float* __restrict__ pooled,
                          const float* __restrict__ fw,
                          const float* __restrict__ fb,
                          float* __restrict__ out) {
  int i = blockIdx.x * blockDim.x + threadIdx.x;
  if (i >= 640) return;
  int b = i / 10, j = i % 10;
  float s = fb[j];
  for (int c = 0; c < 64; ++c) s += pooled[b * 64 + c] * fw[j * 64 + c];
  out[i] = s;
}

// ---------------------------------------------------------------------------
// Host orchestration
// ---------------------------------------------------------------------------
extern "C" void kernel_launch(void* const* d_in, const int* in_sizes, int n_in,
                              void* d_out, int out_size, void* d_ws, size_t ws_size,
                              hipStream_t stream) {
  (void)in_sizes; (void)n_in; (void)out_size; (void)ws_size;

  // Inputs in setup_inputs() dict order
  const float* x      = (const float*)d_in[0];
  const float* conv1w = (const float*)d_in[1];
  const float* conv1b = (const float*)d_in[2];
  struct Res { const float *n1g, *n1b, *n2g, *n2b, *w1, *w2, *wd; };
  Res r1{(const float*)d_in[3], (const float*)d_in[4], (const float*)d_in[5],
         (const float*)d_in[6], (const float*)d_in[7], (const float*)d_in[8],
         (const float*)d_in[9]};
  Res r2{(const float*)d_in[10], (const float*)d_in[11], (const float*)d_in[12],
         (const float*)d_in[13], (const float*)d_in[14], (const float*)d_in[15],
         (const float*)d_in[16]};
  const float* fn1g = (const float*)d_in[17];
  const float* fn1b = (const float*)d_in[18];
  const float* fw1  = (const float*)d_in[19];
  const float* fn2g = (const float*)d_in[20];
  const float* fn2b = (const float*)d_in[21];
  const float* fw2  = (const float*)d_in[22];
  const float* fn3g = (const float*)d_in[23];
  const float* fn3b = (const float*)d_in[24];
  const float* n1g  = (const float*)d_in[25];
  const float* n1b  = (const float*)d_in[26];
  const float* fcw  = (const float*)d_in[27];
  const float* fcb  = (const float*)d_in[28];

  // ---- workspace layout (bytes) ----
  char* W = (char*)d_ws;
  float*          STEM  = (float*)(W + 0);                     // 62,947,328 (reused as ODE pool)
  unsigned short* P1    = (unsigned short*)(W + 62947328);     // 31,473,664 (bf16 pack, reused)
  float*          XDS   = (float*)(W + 94420992);              // 15,736,832
  float*          M1    = (float*)(W + 94420992 + 15736832);   // 15,736,832
  unsigned short* P2    = (unsigned short*)(W + 125894656);    //  7,868,416
  float*          R1OUT = (float*)(W + 133763072);             // 15,736,832
  float*          Y     = (float*)(W + 149499904);             //  4,194,304 (ODE state)
  const size_t G = 153694208;
  unsigned short* R1W1PK = (unsigned short*)(W + G + 0);
  unsigned short* R1W2PK = (unsigned short*)(W + G + 73728);
  unsigned short* R2W1PK = (unsigned short*)(W + G + 147456);
  unsigned short* R2W2PK = (unsigned short*)(W + G + 221184);
  unsigned short* FW1PK  = (unsigned short*)(W + G + 294912);
  unsigned short* FW2PK  = (unsigned short*)(W + G + 368640);
  unsigned short* R1WDPK = (unsigned short*)(W + G + 442368);
  unsigned short* R2WDPK = (unsigned short*)(W + G + 450560);
  float*          FTW    = (float*)(W + G + 458752);
  float*          POOLED = (float*)(W + G + 462848);

  // ODE buffer pool lives inside STEM region (free after P1 is packed)
  char* A = (char*)STEM;
  float* TMP = (float*)(A + 0);
  float* K2  = (float*)(A + 4194304);
  float* K3  = (float*)(A + 8388608);
  float* K4  = (float*)(A + 12582912);
  float* F0  = (float*)(A + 16777216);
  float* F1  = (float*)(A + 20971520);
  float* F2  = (float*)(A + 25165824);
  float* F3  = (float*)(A + 29360128);
  float* FP  = (float*)(A + 33554432);
  float* MID = (float*)(A + 37748736);
  unsigned short* PK  = (unsigned short*)(A + 41943040);
  unsigned short* PK2 = (unsigned short*)(A + 44040192);

  auto gn = [&](const float* in, const float* gg, const float* bb,
                float* oF, unsigned short* oB, int HW, int relu) {
    gn_kernel<<<64 * 32, 128, 0, stream>>>(in, gg, bb, oF, oB, HW, relu);
  };
  auto packw = [&](const float* w, unsigned short* wpk, float* tw, int cin, int taps) {
    int total = taps * 64 * 64;
    pack_w_kernel<<<(total + 255) / 256, 256, 0, stream>>>(w, wpk, tw, cin, taps);
  };
  const int NY = 64 * 16 * 16 * 64;  // 1,048,576
  auto lin = [&](float* o, const float* a0, float c0, const float* a1, float c1,
                 const float* a2, float c2, const float* a3, float c3,
                 const float* a4, float c4) {
    lincomb_kernel<<<(NY + 255) / 256, 256, 0, stream>>>(o, a0, c0, a1, c1,
                                                         a2, c2, a3, c3, a4, c4, NY);
  };

  // Conv grid sizes: ceil(M / 32) with M = 64*Ho*Wo (all divisible by 32)
  const int G31 = (64 * 31 * 31) / 32;   // 1922
  const int G16 = (64 * 16 * 16) / 32;   // 512

  // f(t, src) -> dst  (GN->ReLU->conv(+t ch)->GN->ReLU->conv->GN)
  auto feval = [&](float t, const float* src, float* dst) {
    gn(src, fn1g, fn1b, nullptr, PK, 256, 1);
    conv_wmma_kernel<16,16,16,16,1,1,9><<<G16, 128, 0, stream>>>(
        PK, FW1PK, MID, nullptr, FTW, t);
    gn(MID, fn2g, fn2b, nullptr, PK2, 256, 1);
    conv_wmma_kernel<16,16,16,16,1,1,9><<<G16, 128, 0, stream>>>(
        PK2, FW2PK, MID, nullptr, nullptr, 0.f);
    gn(MID, fn3g, fn3b, dst, nullptr, 256, 0);
  };

  // ---- pack all conv weights to bf16 [tap][co][ci] ----
  packw(r1.w1, R1W1PK, nullptr, 64, 9);
  packw(r1.w2, R1W2PK, nullptr, 64, 9);
  packw(r1.wd, R1WDPK, nullptr, 64, 1);
  packw(r2.w1, R2W1PK, nullptr, 64, 9);
  packw(r2.w2, R2W2PK, nullptr, 64, 9);
  packw(r2.wd, R2WDPK, nullptr, 64, 1);
  packw(fw1,   FW1PK,  FTW,    65, 9);
  packw(fw2,   FW2PK,  nullptr, 64, 9);

  // ---- stem conv ----
  stem_kernel<<<(64 * 62 * 62 * 64) / 256, 256, 0, stream>>>(x, conv1w, conv1b, STEM);

  // ---- residual block 1: 62x62 -> 31x31 ----
  gn(STEM, r1.n1g, r1.n1b, nullptr, P1, 62 * 62, 1);          // STEM free after this
  conv_wmma_kernel<62,62,31,31,2,0,1><<<G31, 128, 0, stream>>>(
      P1, R1WDPK, XDS, nullptr, nullptr, 0.f);
  conv_wmma_kernel<62,62,31,31,2,1,9><<<G31, 128, 0, stream>>>(
      P1, R1W1PK, M1, nullptr, nullptr, 0.f);
  gn(M1, r1.n2g, r1.n2b, nullptr, P2, 31 * 31, 1);
  conv_wmma_kernel<31,31,31,31,1,1,9><<<G31, 128, 0, stream>>>(
      P2, R1W2PK, R1OUT, XDS, nullptr, 0.f);

  // ---- residual block 2: 31x31 -> 16x16 (reuse P1/XDS/M1/P2 regions) ----
  gn(R1OUT, r2.n1g, r2.n1b, nullptr, P1, 31 * 31, 1);
  conv_wmma_kernel<31,31,16,16,2,0,1><<<G16, 128, 0, stream>>>(
      P1, R2WDPK, XDS, nullptr, nullptr, 0.f);
  conv_wmma_kernel<31,31,16,16,2,1,9><<<G16, 128, 0, stream>>>(
      P1, R2W1PK, M1, nullptr, nullptr, 0.f);
  gn(M1, r2.n2g, r2.n2b, nullptr, P2, 16 * 16, 1);
  conv_wmma_kernel<16,16,16,16,1,1,9><<<G16, 128, 0, stream>>>(
      P2, R2W2PK, Y, XDS, nullptr, 0.f);

  // ---- ABM4 ODE integration, t in [0,5], 12 steps ----
  const float dt = 5.0f / 12.0f;
  float* F[4] = {F0, F1, F2, F3};
  feval(0.f, Y, F[0]);
  float t = 0.f;
  for (int i = 0; i < 3; ++i) {                 // RK4 bootstrap (k1 == F[i])
    const float* K1 = F[i];
    lin(TMP, Y, 1.f, K1, 0.5f * dt, nullptr, 0, nullptr, 0, nullptr, 0);
    feval(t + 0.5f * dt, TMP, K2);
    lin(TMP, Y, 1.f, K2, 0.5f * dt, nullptr, 0, nullptr, 0, nullptr, 0);
    feval(t + 0.5f * dt, TMP, K3);
    lin(TMP, Y, 1.f, K3, dt, nullptr, 0, nullptr, 0, nullptr, 0);
    feval(t + dt, TMP, K4);
    lin(Y, Y, 1.f, K1, dt / 6.f, K2, dt / 3.f, K3, dt / 3.f, K4, dt / 6.f);
    t += dt;
    feval(t, Y, F[i + 1]);
  }
  for (int s = 3; s < 12; ++s) {                // AB4 predictor + AM4 corrector
    const float ti = dt * (float)s;
    const float c = dt / 24.f;
    lin(TMP, Y, 1.f, F[3], 55.f * c, F[2], -59.f * c, F[1], 37.f * c, F[0], -9.f * c);
    feval(ti + dt, TMP, FP);
    lin(Y, Y, 1.f, FP, 9.f * c, F[3], 19.f * c, F[2], -5.f * c, F[1], 1.f * c);
    float* old0 = F[0]; F[0] = F[1]; F[1] = F[2]; F[2] = F[3]; F[3] = old0;
    feval(ti + dt, Y, F[3]);
  }

  // ---- head: GN+ReLU -> avgpool -> FC ----
  gn(Y, n1g, n1b, TMP, nullptr, 256, 1);
  pool_kernel<<<64, 64, 0, stream>>>(TMP, POOLED);
  fc_kernel<<<3, 256, 0, stream>>>(POOLED, fcw, fcb, (float*)d_out);
}
